// FlowNet_56916906606924
// MI455X (gfx1250) — matrix operand order repaired and checked by reading, not compiled, not run on hardware
//
#include <hip/hip_runtime.h>
#include <math.h>

// ---------------------------------------------------------------------------
// Types for CDNA5 WMMA (wave32) + TDM descriptor groups
// ---------------------------------------------------------------------------
typedef __attribute__((ext_vector_type(16))) __bf16          v16bf;
typedef __attribute__((ext_vector_type(8)))  float           v8f;
typedef __attribute__((ext_vector_type(8)))  unsigned int    u32x8;
typedef __attribute__((ext_vector_type(4)))  unsigned int    u32x4;
typedef __attribute__((ext_vector_type(8)))  int             i32x8;
typedef __attribute__((ext_vector_type(4)))  int             i32x4;

static __device__ __forceinline__ unsigned short f2bf(float f) {
    // round-to-nearest-even fp32 -> bf16
    unsigned int u = __builtin_bit_cast(unsigned int, f);
    unsigned int r = u + 0x7FFFu + ((u >> 16) & 1u);
    return (unsigned short)(r >> 16);
}

#define GSTRIDE(i, n) \
    for (int i = blockIdx.x * blockDim.x + threadIdx.x; i < (n); i += gridDim.x * blockDim.x)

// ---------------------------------------------------------------------------
// Weight prepack: fp32 OIHW -> bf16 A-fragment order
// layout: [tm][chunk][tap][lane(32)][e(16)]  (one 16x32 bf16 fragment / 512 elems)
//   lane 0-15 : rows m = tm*16 + lane,  K = ck*32 + e
//   lane16-31 : rows m = tm*16 + lane-16, K = ck*32 + 16 + e
// ---------------------------------------------------------------------------
__global__ void prepack_w(const float* __restrict__ Wt, unsigned short* __restrict__ Wp,
                          int Cout, int Cin, int chunks)
{
    int tilesM = (Cout + 15) >> 4;
    int n = tilesM * chunks * 9 * 512;
    GSTRIDE(i, n) {
        int e    = i & 15;
        int lane = (i >> 4) & 31;
        int t    = i >> 9;               // (tm*chunks + ck)*9 + tap
        int tap  = t % 9;
        int rest = t / 9;
        int ck   = rest % chunks;
        int tm   = rest / chunks;
        int m    = tm * 16 + (lane & 15);
        int half = lane >> 4;
        int ka   = ck * 32 + half * 16 + e;
        float v  = 0.0f;
        if (m < Cout && ka < Cin) v = Wt[(m * Cin + ka) * 9 + tap];
        Wp[i] = f2bf(v);
    }
}

// ---------------------------------------------------------------------------
// Tiled implicit-GEMM 3x3 conv (pad=1, stride 1|2), bf16 WMMA.
//  - workgroup: 8x8 output pixels, 8 waves = 2 cout-tiles x 4 pixel-subtiles
//  - A fragments: TDM (tensor_load_to_lds) of prepacked bf16 weights -> LDS
//  - B fragments: cooperative stage of input patch (bf16, [pixel][ch]) -> LDS
//  - inner loop per tap: 2 aligned 32B LDS loads + 1 v_wmma_f32_16x16x32_bf16
// ---------------------------------------------------------------------------
#define CTILE 8

__global__ void __launch_bounds__(256)
conv3x3_wmma_tiled(const float* __restrict__ X, const unsigned short* __restrict__ Wp,
                   const float* __restrict__ Bias, const float* __restrict__ Res,
                   float* __restrict__ Y,
                   int Bn, int Cin, int Hin, int Win,
                   int Cout, int Ho, int Wo, int stride,
                   int chunks, int mGroups, int tilesY, int tilesX,
                   float slope, int doRelu, int addRes)
{
    __shared__ __align__(32) unsigned short ldsA[2 * 9 * 512];   // 18.4 KB
    __shared__ __align__(32) unsigned short ldsB[17 * 17 * 32];  // 18.5 KB

    const int tid  = threadIdx.x;
    const int lane = tid & 31;
    const int wave = tid >> 5;
    const int l16  = lane & 15;
    const int half = lane >> 4;

    // block -> (batch, cout-group, tileY, tileX)
    int bi = blockIdx.x;
    const int tx = bi % tilesX;  bi /= tilesX;
    const int ty = bi % tilesY;  bi /= tilesY;
    const int mg = bi % mGroups; bi /= mGroups;
    const int bb = bi;

    const int tmLocal = wave & 1;
    const int nSub    = wave >> 1;
    const int m0      = (mg * 2 + tmLocal) * 16;
    const bool active = (m0 < Cout);          // wave-uniform

    // this lane's output pixel inside the 8x8 tile (WMMA column)
    const int pid = nSub * 16 + l16;          // 0..63
    const int pty = pid >> 3;
    const int ptx = pid & 7;
    const int ho  = ty * CTILE + pty;
    const int wo  = tx * CTILE + ptx;

    const int ph  = 7 * stride + 3;           // 10 (s=1) or 17 (s=2)
    const int pw  = ph;
    const int pp  = ph * pw;
    const int hiB = ty * CTILE * stride - 1;
    const int wiB = tx * CTILE * stride - 1;
    const int HW  = Hin * Win;

    const int tilesM  = (Cout + 15) >> 4;
    int mtLocal = tilesM - mg * 2; if (mtLocal > 2) mtLocal = 2;

    v8f acc = {};

    for (int ck = 0; ck < chunks; ++ck) {
        __syncthreads();

        // ---- A: TDM load of 2 cout-tiles of prepacked fragments into LDS ----
        if (wave == 0) {
            const unsigned rowDw = 9 * 512 * 2 / 4;                 // 2304 dwords per row
            unsigned long long ga = (unsigned long long)(size_t)(const void*)
                (Wp + ((size_t)(mg * 2) * chunks + ck) * (9 * 512));
            unsigned ldsAddr = (unsigned)(size_t)(void*)&ldsA[0];

            u32x4 g0;
            g0[0] = 1u;                                             // count=1, user D#
            g0[1] = ldsAddr;                                        // lds_addr
            g0[2] = (unsigned)(ga & 0xFFFFFFFFull);                 // global_addr lo
            g0[3] = (unsigned)((ga >> 32) & 0x1FFFFFFull) | (2u << 30); // hi | type=2

            i32x8 g1 = (i32x8)0;
            unsigned td0 = rowDw;                                   // tensor dim0 (dwords)
            unsigned td1 = (unsigned)mtLocal;                       // rows
            unsigned s0  = (unsigned)chunks * rowDw;                // dim0 stride (dwords)
            g1[0] = (int)(2u << 16);                                // data_size = 4B
            g1[1] = (int)((td0 & 0xFFFFu) << 16);                   // tensor_dim0[15:0]
            g1[2] = (int)((td0 >> 16) | ((td1 & 0xFFFFu) << 16));   // td0 hi | td1 lo
            g1[3] = (int)((td1 >> 16) | ((rowDw & 0xFFFFu) << 16)); // td1 hi | tile_dim0
            g1[4] = (int)(td1 & 0xFFFFu);                           // tile_dim1 (tile_dim2=0)
            g1[5] = (int)s0;                                        // dim0 stride lo32
            g1[6] = 0;
            g1[7] = 0;

            i32x4 g2 = (i32x4)0;
            i32x4 g3 = (i32x4)0;
#if defined(__clang_major__) && (__clang_major__ >= 23)
            __builtin_amdgcn_tensor_load_to_lds(g0, g1, g2, g3, (i32x8)0, 0);
#else
            __builtin_amdgcn_tensor_load_to_lds(g0, g1, g2, g3, 0);
#endif
            __builtin_amdgcn_s_wait_tensorcnt(0);
        }

        // ---- B: cooperative patch stage, fp32 NCHW -> bf16 LDS [pixel][ch] ----
        {
            const int cinBase = ck * 32;
            const int patchN  = pp * 32;
            for (int idx = tid; idx < patchN; idx += 256) {
                int c  = idx / pp;               // channel within chunk
                int pl = idx - c * pp;           // pixel-linear (coalesced globally)
                int py = pl / pw;
                int px = pl - py * pw;
                int hi = hiB + py;
                int wi = wiB + px;
                int cc = cinBase + c;
                bool ok = (cc < Cin) && (hi >= 0) && (hi < Hin) && (wi >= 0) && (wi < Win);
                int gidx = ok ? ((bb * Cin + cc) * HW + hi * Win + wi) : 0;
                float v = X[gidx];
                ldsB[pl * 32 + c] = f2bf(ok ? v : 0.0f);
            }
        }
        __syncthreads();

        // ---- GEMM: 9 taps, A and B fragments straight from LDS ----
        if (active) {
            const unsigned short* aBase = ldsA + tmLocal * (9 * 512) + lane * 16;
#pragma unroll
            for (int tap = 0; tap < 9; ++tap) {
                const int ky = tap / 3;
                const int kx = tap - ky * 3;
                const int py = pty * stride + ky;
                const int px = ptx * stride + kx;
                u32x8 araw = *(const u32x8*)(aBase + tap * 512);
                u32x8 braw = *(const u32x8*)(ldsB + (py * pw + px) * 32 + half * 16);
                v16bf av = __builtin_bit_cast(v16bf, araw);
                v16bf bv = __builtin_bit_cast(v16bf, braw);
                acc = __builtin_amdgcn_wmma_f32_16x16x32_bf16(
                          false, av, false, bv, (short)0, acc, false, false);
            }
        }
    }

    // ---- epilogue: bias + optional residual + leaky ReLU, fp32 NCHW ----
    if (active && ho < Ho && wo < Wo) {
#pragma unroll
        for (int r = 0; r < 8; ++r) {
            int co = m0 + half * 8 + r;       // D layout: VGPR r, half selects M / M+8
            if (co < Cout) {
                float v = acc[r] + Bias[co];
                int oi = ((bb * Cout + co) * Ho + ho) * Wo + wo;
                if (addRes) v += Res[oi];
                if (doRelu) v = (v >= 0.0f) ? v : slope * v;
                Y[oi] = v;
            }
        }
    }
}

// ---------------------------------------------------------------------------
// Elementwise / bandwidth-bound kernels
// ---------------------------------------------------------------------------
__global__ void te_kernel(const float* __restrict__ t, float* __restrict__ te,
                          int Bn, int ch)
{
    int n = Bn * ch;
    GSTRIDE(i, n) {
        int c  = i % ch;
        int b  = i / ch;
        int hf = ch >> 1;
        float v = 0.0f;
        if (hf > 0 && c < 2 * hf) {
            int   idx   = (c < hf) ? c : (c - hf);
            int   denom = (hf > 1) ? (hf - 1) : 1;
            float fr    = expf(-logf(10000.0f) * (float)idx / (float)denom);
            float a     = t[b] * fr;
            v = (c < hf) ? sinf(a) : cosf(a);
        }
        te[i] = v;
    }
}

__global__ void semb_kernel(const float* __restrict__ xs, const float* __restrict__ ys,
                            float* __restrict__ s, int n)
{
    GSTRIDE(i, n) s[i] = sinf(10.0f * xs[i]) + sinf(10.0f * ys[i]);
}

__global__ void addmaps_kernel(const float* __restrict__ f, const float* __restrict__ semb,
                               const float* __restrict__ te, float* __restrict__ h,
                               int Bn, int C, int H, int W)
{
    int HW = H * W;
    int n  = Bn * C * HW;
    GSTRIDE(i, n) {
        int hw = i % HW;
        int c  = (i / HW) % C;
        int b  = i / (HW * C);
        h[i] = f[i] + semb[b * HW + hw] + te[b * C + c];
    }
}

__global__ void avgpool_kernel(const float* __restrict__ x, float* __restrict__ y,
                               int Bn, int C, int H, int W)
{
    int Ho = H >> 1, Wo = W >> 1;
    int n  = Bn * C * Ho * Wo;
    GSTRIDE(i, n) {
        int xo = i % Wo;
        int yo = (i / Wo) % Ho;
        int c  = (i / (Wo * Ho)) % C;
        int b  = i / (Wo * Ho * C);
        const float* p = x + ((b * C + c) * H + 2 * yo) * W + 2 * xo;
        y[i] = 0.25f * (p[0] + p[1] + p[W] + p[W + 1]);
    }
}

__global__ void corr_kernel(const float* __restrict__ f1, const float* __restrict__ f2,
                            float* __restrict__ out, int Bn, int C, int H, int W,
                            float slope)
{
    int HW = H * W;
    int n  = Bn * 49 * HW;
    GSTRIDE(i, n) {
        int x  = i % W;
        int y  = (i / W) % H;
        int d  = (i / HW) % 49;
        int b  = i / (HW * 49);
        int dy = d / 7 - 3;
        int dx = d % 7 - 3;
        int ys = y + dy, xs = x + dx;
        float s = 0.0f;
        if (ys >= 0 && ys < H && xs >= 0 && xs < W) {
            const float* a = f1 + (b * C) * HW + y * W + x;
            const float* q = f2 + (b * C) * HW + ys * W + xs;
            for (int c = 0; c < C; ++c) s += a[c * HW] * q[c * HW];
        }
        s /= (float)C;
        out[i] = (s >= 0.0f) ? s : slope * s;
    }
}

__global__ void warp_kernel(const float* __restrict__ img, const float* __restrict__ u,
                            float* __restrict__ out, int Bn, int C, int H, int W,
                            float dt)
{
    int HW = H * W;
    int n  = Bn * HW;
    GSTRIDE(i, n) {
        int x = i % W;
        int y = (i / W) % H;
        int b = i / HW;
        float gx = -1.0f + 2.0f * (float)x / (float)(W - 1);
        float gy = -1.0f + 2.0f * (float)y / (float)(H - 1);
        // NOTE: reference mixes axes: ux from u[:,1]/(H-1)/2, uy from u[:,0]/(W-1)/2
        float ux = u[((b * 2 + 1) * H + y) * W + x] / ((float)(H - 1) * 0.5f);
        float uy = u[((b * 2 + 0) * H + y) * W + x] / ((float)(W - 1) * 0.5f);
        float sx = gx - ux * dt;
        float sy = gy - uy * dt;
        float xf = fminf(fmaxf((sx + 1.0f) * 0.5f * (float)(W - 1), 0.0f), (float)(W - 1));
        float yf = fminf(fmaxf((sy + 1.0f) * 0.5f * (float)(H - 1), 0.0f), (float)(H - 1));
        float x0 = floorf(xf), y0 = floorf(yf);
        float wx = xf - x0,   wy = yf - y0;
        int x0i = (int)x0; x0i = x0i < 0 ? 0 : (x0i > W - 1 ? W - 1 : x0i);
        int y0i = (int)y0; y0i = y0i < 0 ? 0 : (y0i > H - 1 ? H - 1 : y0i);
        int x1i = x0i + 1 > W - 1 ? W - 1 : x0i + 1;
        int y1i = y0i + 1 > H - 1 ? H - 1 : y0i + 1;
        for (int c = 0; c < C; ++c) {
            const float* p = img + (b * C + c) * HW;
            float v00 = p[y0i * W + x0i], v01 = p[y0i * W + x1i];
            float v10 = p[y1i * W + x0i], v11 = p[y1i * W + x1i];
            out[(b * C + c) * HW + y * W + x] =
                v00 * (1 - wx) * (1 - wy) + v01 * wx * (1 - wy) +
                v10 * (1 - wx) * wy       + v11 * wx * wy;
        }
    }
}

// transposed conv 4x4, stride 2, lhs_dilation=2, pad 2, groups=2, flipped weights
__global__ void upflow_kernel(const float* __restrict__ flow, const float* __restrict__ w,
                              float* __restrict__ out, int Bn, int Hin, int Win)
{
    int Ho = Hin * 2, Wo = Win * 2;
    int n  = Bn * 2 * Ho * Wo;
    GSTRIDE(i, n) {
        int xo = i % Wo;
        int yo = (i / Wo) % Ho;
        int g  = (i / (Wo * Ho)) % 2;
        int b  = i / (Wo * Ho * 2);
        float s = 0.0f;
        for (int ky = 0; ky < 4; ++ky) {
            int py = yo + ky - 2;
            if (py < 0 || (py & 1) || (py >> 1) >= Hin) continue;
            for (int kx = 0; kx < 4; ++kx) {
                int px = xo + kx - 2;
                if (px < 0 || (px & 1) || (px >> 1) >= Win) continue;
                s += w[g * 16 + (3 - ky) * 4 + (3 - kx)] *
                     flow[((b * 2 + g) * Hin + (py >> 1)) * Win + (px >> 1)];
            }
        }
        out[i] = s;
    }
}

__global__ void resize_kernel(const float* __restrict__ x, float* __restrict__ y,
                              int Bn, int C, int Hin, int Win, int Ho, int Wo)
{
    int n = Bn * C * Ho * Wo;
    float sh = (float)Hin / (float)Ho;
    float sw = (float)Win / (float)Wo;
    GSTRIDE(i, n) {
        int xo = i % Wo;
        int yo = (i / Wo) % Ho;
        int c  = (i / (Wo * Ho)) % C;
        int b  = i / (Wo * Ho * C);
        float fy = fminf(fmaxf(((float)yo + 0.5f) * sh - 0.5f, 0.0f), (float)(Hin - 1));
        float fx = fminf(fmaxf(((float)xo + 0.5f) * sw - 0.5f, 0.0f), (float)(Win - 1));
        int y0 = (int)floorf(fy);
        int x0 = (int)floorf(fx);
        float wy = fy - (float)y0, wx = fx - (float)x0;
        int y1 = y0 + 1 > Hin - 1 ? Hin - 1 : y0 + 1;
        int x1 = x0 + 1 > Win - 1 ? Win - 1 : x0 + 1;
        const float* p = x + (b * C + c) * Hin * Win;
        y[i] = p[y0 * Win + x0] * (1 - wy) * (1 - wx) + p[y0 * Win + x1] * (1 - wy) * wx +
               p[y1 * Win + x0] * wy * (1 - wx)       + p[y1 * Win + x1] * wy * wx;
    }
}

__global__ void copych_kernel(const float* __restrict__ src, float* __restrict__ dst,
                              int Bn, int Cs, int Cd, int c0, int HW)
{
    int n = Bn * Cs * HW;
    GSTRIDE(i, n) {
        int hw = i % HW;
        int c  = (i / HW) % Cs;
        int b  = i / (HW * Cs);
        dst[(b * Cd + (c0 + c)) * HW + hw] = src[i];
    }
}

// ---------------------------------------------------------------------------
// Host orchestration
// ---------------------------------------------------------------------------
static inline int eg(int n) {
    int b = (n + 255) / 256;
    if (b < 1) b = 1;
    if (b > 16384) b = 16384;
    return b;
}

extern "C" void kernel_launch(void* const* d_in, const int* in_sizes, int n_in,
                              void* d_out, int out_size, void* d_ws, size_t ws_size,
                              hipStream_t stream)
{
    (void)in_sizes; (void)n_in; (void)out_size; (void)ws_size;

    const int Bn = 4;
    const int FN[4] = {32, 64, 96, 128};

    const float* f1 = (const float*)d_in[0];
    const float* f2 = (const float*)d_in[1];
    const float* xg = (const float*)d_in[2];
    const float* yg = (const float*)d_in[3];
    const float* tg = (const float*)d_in[4];

    float* ws = (float*)d_ws;

    // ---- workspace arena (float offsets) ----
    size_t off = 0;
    auto arena = [&](size_t e) { size_t o = off; off += (e + 63) & ~(size_t)63; return o; };

    size_t feat1o[4], feat2o[4], sembo[4];
    for (int l = 0; l < 4; ++l) {
        int Hf = 256 >> l;
        feat1o[l] = arena((size_t)Bn * FN[l] * Hf * Hf);
        feat2o[l] = arena((size_t)Bn * FN[l] * Hf * Hf);
    }
    for (int l = 0; l < 4; ++l) {
        int Hs = 512 >> l;
        sembo[l] = arena((size_t)Bn * Hs * Hs);
    }
    size_t teo    = arena((size_t)Bn * 128);
    size_t wpacko = arena((size_t)80000);                 // packed bf16 weights (<=664KB)
    size_t slabAo = arena((size_t)Bn * 64 * 512 * 512);   // deepest field intermediate
    size_t slabBo = arena((size_t)Bn * 32 * 512 * 512);
    size_t slabCo = arena((size_t)Bn * 66 * 256 * 256);   // corr / block staging
    size_t slabDo = arena((size_t)Bn * 32 * 256 * 256);   // warp / addmaps staging
    size_t fuo    = arena((size_t)Bn * 2 * 256 * 256);    // flow_up
    size_t fmo    = arena((size_t)Bn * 2 * 256 * 256);    // flow_m
    size_t xro    = arena((size_t)Bn * 2 * 512 * 512);    // resized flow

    // ---- conv launcher: prepack weights, then tiled WMMA conv ----
    auto conv = [&](const float* X, const float* W, const float* Bi, const float* Res,
                    float* Y, int Cin, int Hin, int Win, int Cout, int stride,
                    float slope, int relu) {
        int Ho      = (Hin - 1) / stride + 1;
        int Wo      = (Win - 1) / stride + 1;
        int chunks  = (Cin + 31) / 32;
        int tilesM  = (Cout + 15) / 16;
        int mGroups = (tilesM + 1) / 2;
        int tilesY  = (Ho + CTILE - 1) / CTILE;
        int tilesX  = (Wo + CTILE - 1) / CTILE;
        unsigned short* Wp = (unsigned short*)(ws + wpacko);

        prepack_w<<<eg(tilesM * chunks * 9 * 512), 256, 0, stream>>>(W, Wp, Cout, Cin, chunks);

        int blocks = Bn * mGroups * tilesY * tilesX;
        conv3x3_wmma_tiled<<<dim3(blocks), dim3(256), 0, stream>>>(
            X, Wp, Bi, Res, Y, Bn, Cin, Hin, Win, Cout, Ho, Wo, stride,
            chunks, mGroups, tilesY, tilesX, slope, relu, Res != nullptr ? 1 : 0);
    };

    // ---- field layer (conv stack, leaky 0.1 on all but last, residual on last) ----
    auto field = [&](int pbase, const int* chans, int nl, const float* in, int Cin,
                     int H, int W, const float* res, float* out) {
        const float* cur = in;
        int cc = Cin;
        for (int j = 0; j < nl; ++j) {
            const float* w  = (const float*)d_in[pbase + 2 * j];
            const float* bi = (const float*)d_in[pbase + 2 * j + 1];
            if (j == nl - 1) {
                conv(cur, w, bi, res, out, cc, H, W, chans[j], 1, 0.1f, 0);
            } else {
                float* dst = ws + ((j & 1) ? slabBo : slabAo);
                conv(cur, w, bi, nullptr, dst, cc, H, W, chans[j], 1, 0.1f, 1);
                cur = dst;
                cc  = chans[j];
            }
        }
    };

    // ---- semb pyramid: sin(10x)+sin(10y), then avgpool2 per level ----
    semb_kernel<<<eg(Bn * 512 * 512), 256, 0, stream>>>(xg, yg, ws + sembo[0], Bn * 512 * 512);
    for (int l = 0; l < 3; ++l) {
        int Hs = 512 >> l;
        avgpool_kernel<<<eg(Bn * (Hs / 2) * (Hs / 2)), 256, 0, stream>>>(
            ws + sembo[l], ws + sembo[l + 1], Bn, 1, Hs, Hs);
    }

    // ---- feature extraction (params['feat'] flattened at indices 5..20) ----
    auto extract = [&](const float* f0, const size_t* feato) {
        const float* cur = f0;
        int ch = 1;
        for (int l = 0; l < 4; ++l) {
            int Hin = 512 >> l;
            te_kernel<<<eg(Bn * ch), 256, 0, stream>>>(tg, ws + teo, Bn, ch);
            int n = Bn * ch * Hin * Hin;
            addmaps_kernel<<<eg(n), 256, 0, stream>>>(cur, ws + sembo[l], ws + teo,
                                                      ws + slabDo, Bn, ch, Hin, Hin);
            conv(ws + slabDo, (const float*)d_in[5 + 4 * l], (const float*)d_in[6 + 4 * l],
                 nullptr, ws + slabCo, ch, Hin, Hin, FN[l], 2, 0.1f, 1);
            int Hh = Hin >> 1;
            conv(ws + slabCo, (const float*)d_in[7 + 4 * l], (const float*)d_in[8 + 4 * l],
                 nullptr, ws + feato[l], FN[l], Hh, Hh, FN[l], 1, 0.1f, 1);
            cur = ws + feato[l];
            ch  = FN[l];
        }
    };
    extract(f1, feat1o);
    extract(f2, feat2o);

    // ---- output slice offsets (cascaded: level 3,2,1,0, final) ----
    size_t outOff[5];
    {
        size_t o = 0;
        for (int k = 0; k < 4; ++k) {             // level 3 -> 0
            int H = 256 >> (3 - k);
            outOff[k] = o;
            o += (size_t)Bn * 2 * H * H;
        }
        outOff[4] = o;                            // final @512
    }

    const int fieldCh[4] = {128, 64, 32, 2};

    // ---- coarse-to-fine flow cascade ----
    const float* flowPrev = nullptr;
    int Hprev = 0;
    for (int level = 3; level >= 0; --level) {
        int H  = 256 >> level;
        int Cl = FN[level];
        int ub = 21 + 17 * level;                 // unit base: corr(8), flow(8), up_w(1)
        float dt_m = 0.1f / (float)(1 << level);
        float dt_s = dt_m * 0.5f;

        const float* fe1 = ws + feat1o[level];
        const float* fe2 = ws + feat2o[level];
        const float* fe2w;
        const float* corrRes = nullptr;

        if (flowPrev) {
            upflow_kernel<<<eg(Bn * 2 * H * H), 256, 0, stream>>>(
                flowPrev, (const float*)d_in[ub + 16], ws + fuo, Bn, Hprev, Hprev);
            warp_kernel<<<eg(Bn * H * H), 256, 0, stream>>>(
                fe2, ws + fuo, ws + slabDo, Bn, Cl, H, H, -dt_m);
            fe2w    = ws + slabDo;
            corrRes = ws + fuo;
        } else {
            fe2w = fe2;
        }

        corr_kernel<<<eg(Bn * 49 * H * H), 256, 0, stream>>>(
            fe1, fe2w, ws + slabCo, Bn, Cl, H, H, 0.01f);

        // flow_m = flow_up + field(corr)
        field(ub, fieldCh, 4, ws + slabCo, 49, H, H, corrRes, ws + fmo);

        // fe2w2 = project(fe2, flow_m, -dt_s)
        warp_kernel<<<eg(Bn * H * H), 256, 0, stream>>>(
            fe2, ws + fmo, ws + slabDo, Bn, Cl, H, H, -dt_s);

        // block = concat([fe1, fe2w2, flow_m])
        int Cb = 2 * Cl + 2;
        int HW = H * H;
        copych_kernel<<<eg(Bn * Cl * HW), 256, 0, stream>>>(fe1,         ws + slabCo, Bn, Cl, Cb, 0,      HW);
        copych_kernel<<<eg(Bn * Cl * HW), 256, 0, stream>>>(ws + slabDo, ws + slabCo, Bn, Cl, Cb, Cl,     HW);
        copych_kernel<<<eg(Bn * 2  * HW), 256, 0, stream>>>(ws + fmo,    ws + slabCo, Bn, 2,  Cb, 2 * Cl, HW);

        // flow = flow_m + field(block)  -> directly into d_out slice
        float* outp = (float*)d_out + outOff[3 - level];
        field(ub + 8, fieldCh, 4, ws + slabCo, Cb, H, H, ws + fmo, outp);

        flowPrev = outp;
        Hprev    = H;
    }

    // ---- final refinement at 512 ----
    resize_kernel<<<eg(Bn * 2 * 512 * 512), 256, 0, stream>>>(
        flowPrev, ws + xro, Bn, 2, 256, 256, 512, 512);
    int HWf = 512 * 512;
    copych_kernel<<<eg(Bn * 1 * HWf), 256, 0, stream>>>(f1,       ws + slabCo, Bn, 1, 4, 0, HWf);
    copych_kernel<<<eg(Bn * 1 * HWf), 256, 0, stream>>>(f2,       ws + slabCo, Bn, 1, 4, 1, HWf);
    copych_kernel<<<eg(Bn * 2 * HWf), 256, 0, stream>>>(ws + xro, ws + slabCo, Bn, 2, 4, 2, HWf);

    const int upCh[3] = {64, 32, 2};
    field(89, upCh, 3, ws + slabCo, 4, 512, 512, ws + xro, (float*)d_out + outOff[4]);
}
